// NONLocalBlock_72413148610643
// MI455X (gfx1250) — compile-verified, matrix operand order checked by reading, not run on hardware
//
#include <hip/hip_runtime.h>
#include <math.h>

// Problem constants (from reference): B=8, C=512, CI=128, H=W=64
#define CC   512
#define CIc  128
#define Bc   8
#define Hc   64
#define Wc   64
#define Nc   4096   // H*W
#define Mc   1024   // N/4 after 2x2 maxpool

typedef __attribute__((ext_vector_type(16))) __bf16        v16bf;
typedef __attribute__((ext_vector_type(8)))  float         v8f;
typedef __attribute__((ext_vector_type(4)))  unsigned int  u32x4;

union V16U { v16bf v; u32x4 q[2]; };

__device__ __forceinline__ unsigned short f2bf(float f) {
    unsigned u = __float_as_uint(f);
    u += 0x7FFFu + ((u >> 16) & 1u);     // round-to-nearest-even
    return (unsigned short)(u >> 16);
}
__device__ __forceinline__ float bf2f(unsigned short s) {
    return __uint_as_float(((unsigned)s) << 16);
}
// Load one WMMA 16-bit operand for this lane: two contiguous 16B chunks
// (CDNA5 16-bit operand layout: VGPR0-3 = chunk0, VGPR4-7 = chunk1).
__device__ __forceinline__ v16bf ld_ab(const unsigned short* p0, const unsigned short* p1) {
    V16U u;
    u.q[0] = *(const u32x4*)p0;
    u.q[1] = *(const u32x4*)p1;
    return u.v;
}
__device__ __forceinline__ v8f wmma_bf16(v16bf a, v16bf b, v8f c) {
    // D = A(16x32) * B(32x16) + C, f32 accumulate
    return __builtin_amdgcn_wmma_f32_16x16x32_bf16(false, a, false, b, (short)0, c, false, false);
}
__device__ __forceinline__ v8f vzero() {
    v8f r;
#pragma unroll
    for (int i = 0; i < 8; i++) r[i] = 0.f;
    return r;
}

// ---- CDNA5 async global->LDS DMA (ASYNCcnt-tracked, no VGPR round-trip) ----
// Generic pointer to __shared__ carries the LDS byte offset in its low 32 bits
// (flat aperture mapping).
__device__ __forceinline__ void async_cp16(void* lds, const void* g) {
    unsigned l = (unsigned)(size_t)lds;
    asm volatile("global_load_async_to_lds_b128 %0, %1, off"
                 :: "v"(l), "v"(g)
                 : "memory");
}
__device__ __forceinline__ void wait_async0() {
    asm volatile("s_wait_asynccnt 0" ::: "memory");
}
__device__ __forceinline__ void wait_async4() {   // allow next tile (4 ops) in flight
    asm volatile("s_wait_asynccnt 4" ::: "memory");
}

// ---------------------------------------------------------------------------
// K0a: convert the four weight matrices to bf16 (layouts preserved:
// g/phi/theta: [CI][C] row-major (K=C contiguous), W: [C][CI] (K=CI contiguous))
__global__ void k_cvt_weights(const float* __restrict__ gw, const float* __restrict__ pw,
                              const float* __restrict__ tw, const float* __restrict__ Ww,
                              unsigned short* gwb, unsigned short* pwb,
                              unsigned short* twb, unsigned short* Wwb) {
    int i = blockIdx.x * 256 + threadIdx.x;   // 65536 elements each
    gwb[i] = f2bf(gw[i]);
    pwb[i] = f2bf(pw[i]);
    twb[i] = f2bf(tw[i]);
    Wwb[i] = f2bf(Ww[i]);
}

// ---------------------------------------------------------------------------
// K0b: tiled transpose+convert  x[b][c][n] f32  ->  x_t[b][n][c] bf16
__global__ void k_xt(const float* __restrict__ x, unsigned short* __restrict__ xt) {
    __shared__ float tile[32][33];
    const int b  = blockIdx.z;
    const int n0 = blockIdx.x * 32;
    const int c0 = blockIdx.y * 32;
    const float* xp = x + ((size_t)b * CC) * Nc;
#pragma unroll
    for (int i = threadIdx.y; i < 32; i += 8)
        tile[i][threadIdx.x] = xp[(size_t)(c0 + i) * Nc + n0 + threadIdx.x];
    __syncthreads();
    unsigned short* op = xt + ((size_t)b * Nc) * CC;
#pragma unroll
    for (int i = threadIdx.y; i < 32; i += 8)
        op[(size_t)(n0 + i) * CC + c0 + threadIdx.x] = f2bf(tile[threadIdx.x][i]);
}

// ---------------------------------------------------------------------------
// K1: the three projection GEMMs. Grid = (N/16, B), 8 waves/WG.
// The shared A tile (16 rows x 512 K, 16KB contiguous) is DMA'd into LDS once
// per WG; each wave computes ci-tile w of theta/phi/g (weights B from global).
__global__ __launch_bounds__(256) void k_proj(
        const unsigned short* __restrict__ xt,
        const unsigned short* __restrict__ twb, const unsigned short* __restrict__ pwb,
        const unsigned short* __restrict__ gwb,
        const float* __restrict__ tb, const float* __restrict__ pb, const float* __restrict__ gb,
        unsigned short* __restrict__ theta_t, unsigned short* __restrict__ phi_f,
        unsigned short* __restrict__ g_f) {
    __shared__ __align__(16) unsigned short As[16 * CC];   // 16 KB
    const int b     = blockIdx.y;
    const int nBase = blockIdx.x * 16;
    const int wave  = threadIdx.x >> 5;
    const int lane  = threadIdx.x & 31;
    const int col   = lane & 15;
    const int h     = (lane >> 4) & 1;
    const int t     = wave;                       // ci tile 0..7

    // stage A tile: contiguous 16KB = 1024 x 16B chunks, 4 per thread
    {
        const char* aSrc = (const char*)(xt + ((size_t)b * Nc + nBase) * CC);
        char* aDst = (char*)As;
#pragma unroll
        for (int k = 0; k < 4; k++) {
            const unsigned o = (threadIdx.x * 4u + k) * 16u;
            async_cp16(aDst + o, aSrc + o);
        }
        wait_async0();
        __syncthreads();
    }

    const unsigned short* arow = As  + col * CC;                 // LDS A rows
    const unsigned short* trow = twb + (size_t)(t * 16 + col) * CC;
    const unsigned short* prow = pwb + (size_t)(t * 16 + col) * CC;
    const unsigned short* grow = gwb + (size_t)(t * 16 + col) * CC;

    v8f aT = vzero(), aP = vzero(), aG = vzero();
#pragma unroll
    for (int kk = 0; kk < 16; kk++) {            // K = 512, step 32
        const int cb = kk * 32;
        v16bf A  = ld_ab(arow + cb + h * 8, arow + cb + 16 + h * 8);   // ds_load_b128
        v16bf Bt = ld_ab(trow + cb + h * 8, trow + cb + 16 + h * 8);
        v16bf Bp = ld_ab(prow + cb + h * 8, prow + cb + 16 + h * 8);
        v16bf Bg = ld_ab(grow + cb + h * 8, grow + cb + 16 + h * 8);
        aT = wmma_bf16(A, Bt, aT);
        aP = wmma_bf16(A, Bp, aP);
        aG = wmma_bf16(A, Bg, aG);
    }
    const float bT = tb[t * 16 + col], bP = pb[t * 16 + col], bG = gb[t * 16 + col];
#pragma unroll
    for (int j = 0; j < 8; j++) {                // D layout: row = j + 8h, col = lane%16
        const int n = nBase + j + 8 * h;
        const size_t o = ((size_t)b * Nc + n) * CIc + t * 16 + col;
        theta_t[o] = f2bf(aT[j] + bT);
        phi_f[o]   = f2bf(aP[j] + bP);
        g_f[o]     = f2bf(aG[j] + bG);
    }
}

// ---------------------------------------------------------------------------
// K2: 2x2 maxpool. phi -> phi_p[b][m][ci] (K-contig for B operand),
//                  g   -> g_pT[b][ci][m]  (m-contig: column-major B for attn*V)
__global__ void k_pool(const unsigned short* __restrict__ phi_f,
                       const unsigned short* __restrict__ g_f,
                       unsigned short* __restrict__ phi_p,
                       unsigned short* __restrict__ g_pT) {
    const int idx = blockIdx.x * 256 + threadIdx.x;   // B*M*CI = 1,048,576
    const int ci = idx & (CIc - 1);
    const int m  = (idx >> 7) & (Mc - 1);
    const int b  = idx >> 17;
    const int hm = m >> 5, wm = m & 31;
    const int n00 = (hm * 2) * Wc + wm * 2;
    const size_t base = ((size_t)b * Nc) * CIc + ci;
    const size_t o00 = base + (size_t)n00 * CIc;
    const size_t o01 = o00 + CIc;
    const size_t o10 = o00 + (size_t)Wc * CIc;
    const size_t o11 = o10 + CIc;

    float pm = fmaxf(fmaxf(bf2f(phi_f[o00]), bf2f(phi_f[o01])),
                     fmaxf(bf2f(phi_f[o10]), bf2f(phi_f[o11])));
    float gm = fmaxf(fmaxf(bf2f(g_f[o00]),  bf2f(g_f[o01])),
                     fmaxf(bf2f(g_f[o10]),  bf2f(g_f[o11])));
    phi_p[((size_t)b * Mc + m) * CIc + ci] = f2bf(pm);
    g_pT [((size_t)b * CIc + ci) * Mc + m] = f2bf(gm);
}

// ---------------------------------------------------------------------------
// K3: flash-style attention. Grid = (N/128, B), 8 waves/WG, each wave owns a
// 16-row query tile. phi/g key-value tiles are shared by all 8 waves, so they
// are double-buffered in LDS via async DMA (4 ops/tile/wave on ASYNCcnt);
// scores stay in registers; P re-layout D->A uses a wave-private LDS slice.
__global__ __launch_bounds__(256) void k_attn(
        const unsigned short* __restrict__ theta_t,
        const unsigned short* __restrict__ phi_p,
        const unsigned short* __restrict__ g_pT,
        unsigned short* __restrict__ y_t) {
    __shared__ __align__(16) unsigned short Pb[8][16 * 32];     //  8 KB
    __shared__ __align__(16) unsigned short phiS[2][32 * CIc];  // 16 KB
    __shared__ __align__(16) unsigned short gS[2][CIc * 32];    // 16 KB
    const int b    = blockIdx.y;
    const int wave = threadIdx.x >> 5;
    const int lane = threadIdx.x & 31;
    const int col  = lane & 15;
    const int h    = (lane >> 4) & 1;
    const int tid  = threadIdx.x;
    const int nBase = blockIdx.x * 128 + wave * 16;

    const char* phSrc = (const char*)(phi_p + ((size_t)b * Mc) * CIc);
    const char* gSrc  = (const char*)(g_pT  + ((size_t)b * CIc) * Mc);

    // stage one 32-wide m tile: phi contiguous 8KB, g = 128 rows x 64B
    auto stage = [&](int buf, int mb) {
        char* pd = (char*)&phiS[buf][0];
        const char* ps = phSrc + (size_t)mb * CIc * 2;
        const unsigned o = tid * 32u;
        async_cp16(pd + o, ps + o);
        async_cp16(pd + o + 16, ps + o + 16);
        char* gd = (char*)&gS[buf][0];
#pragma unroll
        for (int k = 0; k < 2; k++) {
            const int c   = tid * 2 + k;       // 512 chunks of 16B
            const int row = c >> 2;
            const int oo  = (c & 3) * 16;
            async_cp16(gd + row * 64 + oo,
                       gSrc + (size_t)row * (Mc * 2) + (size_t)mb * 2 + oo);
        }
    };

    // Q rows resident for the whole M loop (K = CI = 128 -> 4 chunks of 32)
    const unsigned short* qrow = theta_t + ((size_t)b * Nc + nBase + col) * CIc;
    v16bf A[4];
#pragma unroll
    for (int kk = 0; kk < 4; kk++)
        A[kk] = ld_ab(qrow + kk * 32 + h * 8, qrow + kk * 32 + 16 + h * 8);

    v8f acc[8];
#pragma unroll
    for (int t = 0; t < 8; t++) acc[t] = vzero();
    float mrow[8], lrow[8];
#pragma unroll
    for (int j = 0; j < 8; j++) { mrow[j] = -3.0e38f; lrow[j] = 0.f; }

    stage(0, 0);
    unsigned short* Pw = &Pb[wave][0];

    for (int i = 0; i < Mc / 32; i++) {
        const int p = i & 1;
        if (i + 1 < Mc / 32) { stage(1 - p, (i + 1) * 32); wait_async4(); }
        else                 { wait_async0(); }
        __syncthreads();

        // S tiles: two 16x16 score tiles (m and m+16), K = 128, B from LDS
        v8f s0 = vzero(), s1 = vzero();
        const unsigned short* lp0 = &phiS[p][col * CIc];
        const unsigned short* lp1 = &phiS[p][(16 + col) * CIc];
#pragma unroll
        for (int kk = 0; kk < 4; kk++) {
            v16bf B0 = ld_ab(lp0 + kk * 32 + h * 8, lp0 + kk * 32 + 16 + h * 8);
            v16bf B1 = ld_ab(lp1 + kk * 32 + h * 8, lp1 + kk * 32 + 16 + h * 8);
            s0 = wmma_bf16(A[kk], B0, s0);
            s1 = wmma_bf16(A[kk], B1, s1);
        }
        // online softmax per row (rows striped: VGPR j -> row j + 8h, col = lane%16)
#pragma unroll
        for (int j = 0; j < 8; j++) {
            float a0 = s0[j], a1 = s1[j];
            float tm = fmaxf(a0, a1);
#pragma unroll
            for (int k = 1; k < 16; k <<= 1) tm = fmaxf(tm, __shfl_xor(tm, k, 32));
            const float mn  = fmaxf(mrow[j], tm);
            const float sc  = __expf(mrow[j] - mn);
            const float e0  = __expf(a0 - mn);
            const float e1  = __expf(a1 - mn);
            float rs = e0 + e1;
#pragma unroll
            for (int k = 1; k < 16; k <<= 1) rs += __shfl_xor(rs, k, 32);
            lrow[j] = lrow[j] * sc + rs;
            mrow[j] = mn;
#pragma unroll
            for (int t = 0; t < 8; t++) acc[t][j] *= sc;
            const int row = j + 8 * h;
            Pw[row * 32 + col]      = f2bf(e0);
            Pw[row * 32 + 16 + col] = f2bf(e1);
        }
        // Re-read P in A-operand layout (16 rows x K=32 of m)
        const unsigned short* pr = Pw + col * 32;
        v16bf AP = ld_ab(pr + h * 8, pr + 16 + h * 8);
        // y[n][ci] += P * g^T   (B = gS rows [ci][m], m-contiguous)
#pragma unroll
        for (int t = 0; t < 8; t++) {
            const unsigned short* gr = &gS[p][(t * 16 + col) * 32];
            v16bf Bg = ld_ab(gr + h * 8, gr + 16 + h * 8);
            acc[t] = wmma_bf16(AP, Bg, acc[t]);
        }
        __syncthreads();
    }
    // finalize: divide by row sums, emit y_t[b][n][ci] bf16 (K-contig for K4's B)
#pragma unroll
    for (int j = 0; j < 8; j++) {
        const float inv = 1.0f / lrow[j];
        const int n = nBase + j + 8 * h;
        unsigned short* o = y_t + ((size_t)b * Nc + n) * CIc + col;
#pragma unroll
        for (int t = 0; t < 8; t++) o[t * 16] = f2bf(acc[t][j] * inv);
    }
}

// ---------------------------------------------------------------------------
// K4: W projection (K=128) + channels-first LayerNorm + residual.
// Grid = (N/16, B); WG covers all 512 out channels for one 16-col n-tile,
// each of 8 waves owns 4 c-tiles. The shared y B-operand tile (4KB) is DMA'd
// into LDS once. Deterministic LN reduction via LDS partials.
__global__ __launch_bounds__(256) void k_out(
        const unsigned short* __restrict__ Wwb, const unsigned short* __restrict__ y_t,
        const float* __restrict__ Wb, const float* __restrict__ gam,
        const float* __restrict__ bet, const float* __restrict__ x,
        float* __restrict__ out) {
    __shared__ __align__(16) unsigned short yS[16 * CIc];   // 4 KB
    __shared__ float ps[8][2][16];
    __shared__ float qs[8][2][16];
    const int b     = blockIdx.y;
    const int nBase = blockIdx.x * 16;
    const int wave  = threadIdx.x >> 5;
    const int lane  = threadIdx.x & 31;
    const int col   = lane & 15;
    const int h     = (lane >> 4) & 1;
    const int cBase = wave * 64;

    // stage y tile: contiguous 4KB = 256 x 16B chunks, 1 per thread
    {
        const char* ySrc = (const char*)(y_t + ((size_t)b * Nc + nBase) * CIc);
        async_cp16((char*)yS + threadIdx.x * 16u, ySrc + threadIdx.x * 16u);
        wait_async0();
        __syncthreads();
    }

    // shared B operand: y columns (n = nBase + col), K = ci contiguous (LDS)
    const unsigned short* yr = yS + col * CIc;
    v16bf By[4];
#pragma unroll
    for (int kk = 0; kk < 4; kk++)
        By[kk] = ld_ab(yr + kk * 32 + h * 8, yr + kk * 32 + 16 + h * 8);

    v8f acc[4];
#pragma unroll
    for (int t = 0; t < 4; t++) acc[t] = vzero();
#pragma unroll
    for (int t = 0; t < 4; t++) {
        const unsigned short* wr = Wwb + (size_t)(cBase + t * 16 + col) * CIc;
#pragma unroll
        for (int kk = 0; kk < 4; kk++) {
            v16bf Aw = ld_ab(wr + kk * 32 + h * 8, wr + kk * 32 + 16 + h * 8);
            acc[t] = wmma_bf16(Aw, By[kk], acc[t]);
        }
    }
    // bias + per-lane LN partials (this lane covers 32 c values of column n=col)
    float lsum = 0.f, lsq = 0.f;
#pragma unroll
    for (int t = 0; t < 4; t++) {
#pragma unroll
        for (int j = 0; j < 8; j++) {
            const int c = cBase + t * 16 + j + 8 * h;
            const float v = acc[t][j] + Wb[c];
            acc[t][j] = v;
            lsum += v;
            lsq  += v * v;
        }
    }
    ps[wave][h][col] = lsum;
    qs[wave][h][col] = lsq;
    __syncthreads();
    float sum = 0.f, sq = 0.f;
#pragma unroll
    for (int w = 0; w < 8; w++) {
        sum += ps[w][0][col] + ps[w][1][col];
        sq  += qs[w][0][col] + qs[w][1][col];
    }
    const float mean = sum * (1.0f / CC);
    const float var  = sq * (1.0f / CC) - mean * mean;
    const float rstd = rsqrtf(var + 1e-5f);
    const int n = nBase + col;
#pragma unroll
    for (int t = 0; t < 4; t++) {
#pragma unroll
        for (int j = 0; j < 8; j++) {
            const int c = cBase + t * 16 + j + 8 * h;
            const size_t o = ((size_t)b * CC + c) * Nc + n;
            out[o] = (acc[t][j] - mean) * rstd * gam[c] + bet[c] + x[o];
        }
    }
}

// ---------------------------------------------------------------------------
extern "C" void kernel_launch(void* const* d_in, const int* in_sizes, int n_in,
                              void* d_out, int out_size, void* d_ws, size_t ws_size,
                              hipStream_t stream) {
    (void)in_sizes; (void)n_in; (void)out_size; (void)ws_size;
    const float* x       = (const float*)d_in[0];
    const float* g_w     = (const float*)d_in[1];
    const float* g_b     = (const float*)d_in[2];
    const float* phi_w   = (const float*)d_in[3];
    const float* phi_b   = (const float*)d_in[4];
    const float* theta_w = (const float*)d_in[5];
    const float* theta_b = (const float*)d_in[6];
    const float* W_w     = (const float*)d_in[7];
    const float* W_b     = (const float*)d_in[8];
    const float* ln_g    = (const float*)d_in[9];
    const float* ln_b    = (const float*)d_in[10];
    float* out = (float*)d_out;

    char* ws = (char*)d_ws;
    // workspace layout (all offsets 16B-aligned)
    unsigned short* xt      = (unsigned short*)(ws);                 // 32 MB (reused as y_t)
    unsigned short* theta_t = (unsigned short*)(ws + 33554432);      //  8 MB
    unsigned short* phi_f   = (unsigned short*)(ws + 41943040);      //  8 MB
    unsigned short* g_f     = (unsigned short*)(ws + 50331648);      //  8 MB
    unsigned short* phi_p   = (unsigned short*)(ws + 58720256);      //  2 MB
    unsigned short* g_pT    = (unsigned short*)(ws + 60817408);      //  2 MB
    unsigned short* twb     = (unsigned short*)(ws + 62914560);
    unsigned short* pwb     = (unsigned short*)(ws + 62914560 + 131072);
    unsigned short* gwb     = (unsigned short*)(ws + 62914560 + 262144);
    unsigned short* Wwb     = (unsigned short*)(ws + 62914560 + 393216);
    unsigned short* y_t     = xt;   // xt dead after k_proj; reuse for y

    k_cvt_weights<<<256, 256, 0, stream>>>(g_w, phi_w, theta_w, W_w, gwb, pwb, twb, Wwb);
    k_xt<<<dim3(Nc / 32, CC / 32, Bc), dim3(32, 8), 0, stream>>>(x, xt);
    k_proj<<<dim3(Nc / 16, Bc), 256, 0, stream>>>(xt, twb, pwb, gwb,
                                                  theta_b, phi_b, g_b,
                                                  theta_t, phi_f, g_f);
    k_pool<<<(Bc * Mc * CIc) / 256, 256, 0, stream>>>(phi_f, g_f, phi_p, g_pT);
    k_attn<<<dim3(Nc / 128, Bc), 256, 0, stream>>>(theta_t, phi_p, g_pT, y_t);
    k_out<<<dim3(Nc / 16, Bc), 256, 0, stream>>>(Wwb, y_t, W_b, ln_g, ln_b, x, out);
}
